// EpicPred_15839839388168
// MI455X (gfx1250) — matrix operand order repaired and checked by reading, not compiled, not run on hardware
//
#include <hip/hip_runtime.h>
#include <hip/hip_bf16.h>

#define C_NUM 16
#define G_NUM 2048
#define N_NUM (C_NUM * G_NUM)   // 32768
#define D_IN  768
#define E_DIM 64
#define H_NUM 4
#define DH    16

typedef _Float16 h8   __attribute__((ext_vector_type(8)));
typedef _Float16 v16h __attribute__((ext_vector_type(16)));
typedef float    v8f  __attribute__((ext_vector_type(8)));
typedef unsigned int u32x4 __attribute__((ext_vector_type(4)));
typedef int          i32x8 __attribute__((ext_vector_type(8)));
typedef int          i32x4 __attribute__((ext_vector_type(4)));

__device__ __forceinline__ v8f vzero8f() {
  v8f z;
  #pragma unroll
  for (int i = 0; i < 8; ++i) z[i] = 0.f;
  return z;
}
__device__ __forceinline__ h8 hzero8() {
  h8 z;
  #pragma unroll
  for (int i = 0; i < 8; ++i) z[i] = (_Float16)0.f;
  return z;
}
__device__ __forceinline__ v16h hzero16() {
  v16h z;
  #pragma unroll
  for (int i = 0; i < 16; ++i) z[i] = (_Float16)0.f;
  return z;
}
__device__ __forceinline__ v16h cat16(h8 a, h8 b) {
  return __builtin_shufflevector(a, b, 0,1,2,3,4,5,6,7,8,9,10,11,12,13,14,15);
}
__device__ __forceinline__ v8f wmma16(v16h a, v16h b, v8f c) {
  return __builtin_amdgcn_wmma_f32_16x16x32_f16(false, a, false, b, (short)0, c, false, false);
}
__device__ __forceinline__ float redmax16(float v) {
  #pragma unroll
  for (int m = 1; m < 16; m <<= 1) v = fmaxf(v, __shfl_xor(v, m, 32));
  return v;
}
__device__ __forceinline__ float redsum16(float v) {
  #pragma unroll
  for (int m = 1; m < 16; m <<= 1) v += __shfl_xor(v, m, 32);
  return v;
}

// ------------------------------------------------ Tensor Data Mover helper
// 2D f16 tile load: global (row_stride_elems per row) -> dense LDS tile.
// D# layout per CDNA5 ISA ch.8 (group0: count/lds/global/type; group1: dims).
// This toolchain's builtin takes 6 args (g0, g1, g2, g3, g4, cpol).
__device__ __forceinline__ void tdm_load_2d_f16(unsigned lds_off, const void* gptr,
                                                unsigned tile_d0, unsigned tile_d1,
                                                unsigned row_stride_elems) {
  const unsigned long long ga = (unsigned long long)(uintptr_t)gptr;
  u32x4 g0;
  g0[0] = 1u;                                               // count=1, user D#
  g0[1] = lds_off;                                          // lds_addr (bytes)
  g0[2] = (unsigned)ga;                                     // global_addr[31:0]
  g0[3] = (unsigned)((ga >> 32) & 0x1FFFFFFu) | (2u << 30); // addr[56:32] | type=2
  i32x8 g1;
  g1[0] = (int)(1u << 16);                                  // data_size=1 (2 bytes)
  g1[1] = (int)((row_stride_elems & 0xFFFFu) << 16);        // tensor_dim0[15:0]
  g1[2] = (int)(((row_stride_elems >> 16) & 0xFFFFu) |
                ((tile_d1 & 0xFFFFu) << 16));               // dim0[31:16] | tensor_dim1[15:0]
  g1[3] = (int)(((tile_d1 >> 16) & 0xFFFFu) |
                ((tile_d0 & 0xFFFFu) << 16));               // tensor_dim1[31:16] | tile_dim0
  g1[4] = (int)(tile_d1 & 0xFFFFu);                         // tile_dim1 (tile_dim2=0)
  g1[5] = (int)row_stride_elems;                            // tensor_dim0_stride[31:0]
  g1[6] = 0;
  g1[7] = 0;
  i32x4 gz;
  gz[0] = 0; gz[1] = 0; gz[2] = 0; gz[3] = 0;
  i32x8 gz8;
  #pragma unroll
  for (int i = 0; i < 8; ++i) gz8[i] = 0;
  __builtin_amdgcn_tensor_load_to_lds(g0, g1, gz, gz, gz8, 0);
}

// ---------------------------------------------------------------- grouping
// stable argsort by label (equal-size clusters): one wave per cluster.
__global__ void order_kernel(const int* __restrict__ label,
                             const int* __restrict__ barcode,
                             int* __restrict__ ord,
                             float* __restrict__ bcs_out) {
  const int c = blockIdx.x;
  const int lane = threadIdx.x;
  int running = 0;
  for (int base = 0; base < N_NUM; base += 32) {
    const int i = base + lane;
    const bool m = (label[i] == c);
    const unsigned mask = (unsigned)__ballot(m);
    if (m) {
      const int r = running + __popc(mask & ((1u << lane) - 1u));
      ord[c * G_NUM + r] = i;
      bcs_out[c * G_NUM + r] = (float)barcode[i];
    }
    running += __popc(mask);
  }
}

// ---------------------------------------------------------- weight prep
// WeT[n][k] = (f16)We[k][n]; Wt[n][k] = (f16)W[k][n] for the 4 MHA1 mats.
__global__ void prep_kernel(const float* __restrict__ We,
                            const float* __restrict__ Wq, const float* __restrict__ Wk,
                            const float* __restrict__ Wv, const float* __restrict__ Wo,
                            _Float16* __restrict__ WeT,
                            _Float16* __restrict__ Wqt, _Float16* __restrict__ Wkt,
                            _Float16* __restrict__ Wvt, _Float16* __restrict__ Wot) {
  const int idx = blockIdx.x * blockDim.x + threadIdx.x;
  if (idx < D_IN * E_DIM) {                     // 49152
    const int n = idx / D_IN, k = idx % D_IN;
    WeT[n * D_IN + k] = (_Float16)We[k * E_DIM + n];
  } else {
    int i2 = idx - D_IN * E_DIM;
    const int mat = i2 >> 12;                   // 0..3
    const int o = i2 & 4095;
    const int n = o >> 6, k = o & 63;
    const float* src = (mat == 0) ? Wq : (mat == 1) ? Wk : (mat == 2) ? Wv : Wo;
    _Float16* dst = (mat == 0) ? Wqt : (mat == 1) ? Wkt : (mat == 2) ? Wvt : Wot;
    dst[n * 64 + k] = (_Float16)src[k * 64 + n];
  }
}

// ---------------------------------------------------------- embedding GEMM
// emb = relu(X[ord] @ We + be), f16 out. One wave per 16-row tile; the
// workgroup stages WeT in LDS via the Tensor Data Mover in two 48KB phases.
#define KCHUNK 384   // K elems per phase (2 phases x 384 = 768)
__global__ __launch_bounds__(256) void embed_kernel(const float* __restrict__ x,
                                                    const int* __restrict__ ord,
                                                    const float* __restrict__ be,
                                                    const _Float16* __restrict__ WeT,
                                                    _Float16* __restrict__ embH) {
  __shared__ __align__(16) _Float16 Bs[E_DIM * KCHUNK];  // 48KB weight panel
  const int tile = blockIdx.x * (blockDim.x >> 5) + (threadIdx.x >> 5);
  const int lane = threadIdx.x & 31;
  const int n0 = lane & 15;
  const int hi = lane >> 4;
  const int inst = ord[tile * 16 + n0];
  const float* xr = x + (size_t)inst * D_IN;

  v8f acc[4];
  #pragma unroll
  for (int nt = 0; nt < 4; ++nt) acc[nt] = vzero8f();

  for (int ph = 0; ph < 2; ++ph) {
    if (threadIdx.x == 0) {
      // WeT chunk: 64 rows (cols of We) x 384 halfs, row stride 768 halfs.
      tdm_load_2d_f16((unsigned)(uintptr_t)(void*)Bs,
                      WeT + ph * KCHUNK, KCHUNK, E_DIM, D_IN);
    }
    if (threadIdx.x < 32) __builtin_amdgcn_s_wait_tensorcnt(0);
    __syncthreads();

    for (int ks = 0; ks < KCHUNK / 32; ++ks) {
      const int kA = ph * KCHUNK + ks * 32 + hi * 8;
      const float4 f0 = *(const float4*)(xr + kA);
      const float4 f1 = *(const float4*)(xr + kA + 4);
      const float4 f2 = *(const float4*)(xr + kA + 16);
      const float4 f3 = *(const float4*)(xr + kA + 20);
      v16h a;
      a[0]=(_Float16)f0.x; a[1]=(_Float16)f0.y; a[2]=(_Float16)f0.z; a[3]=(_Float16)f0.w;
      a[4]=(_Float16)f1.x; a[5]=(_Float16)f1.y; a[6]=(_Float16)f1.z; a[7]=(_Float16)f1.w;
      a[8]=(_Float16)f2.x; a[9]=(_Float16)f2.y; a[10]=(_Float16)f2.z; a[11]=(_Float16)f2.w;
      a[12]=(_Float16)f3.x; a[13]=(_Float16)f3.y; a[14]=(_Float16)f3.z; a[15]=(_Float16)f3.w;

      #pragma unroll
      for (int nt = 0; nt < 4; ++nt) {
        const int col = nt * 16 + n0;
        const _Float16* wp = Bs + col * KCHUNK + ks * 32 + hi * 8;
        const v16h b = cat16(*(const h8*)wp, *(const h8*)(wp + 16));
        acc[nt] = wmma16(a, b, acc[nt]);
      }
    }
    __syncthreads();   // protect Bs before next-phase TDM overwrite
  }

  #pragma unroll
  for (int nt = 0; nt < 4; ++nt) {
    const int col = nt * 16 + n0;
    const float bb = be[col];
    #pragma unroll
    for (int r = 0; r < 8; ++r) {
      const int row = tile * 16 + r + 8 * hi;
      float v = acc[nt][r] + bb;
      v = fmaxf(v, 0.f);
      embH[(size_t)row * 64 + col] = (_Float16)v;
    }
  }
}

// ------------------------------------------------- generic 64x64 projection
// Y = X @ W + b (optionally scaled). MODE 0: row-major f16. MODE 1: transposed.
// Weight (8KB) staged in LDS by the Tensor Data Mover once per workgroup.
template <int MODE>
__global__ __launch_bounds__(256) void proj64_kernel(const _Float16* __restrict__ Xh,
                                                     const _Float16* __restrict__ Wt,
                                                     const float* __restrict__ bias,
                                                     float scale,
                                                     _Float16* __restrict__ out) {
  __shared__ __align__(16) _Float16 Ws[64 * 64];
  if (threadIdx.x == 0) {
    tdm_load_2d_f16((unsigned)(uintptr_t)(void*)Ws, Wt, 4096, 1, 4096);
  }
  if (threadIdx.x < 32) __builtin_amdgcn_s_wait_tensorcnt(0);
  __syncthreads();

  const int tile = blockIdx.x * (blockDim.x >> 5) + (threadIdx.x >> 5);
  const int lane = threadIdx.x & 31;
  const int n0 = lane & 15;
  const int hi = lane >> 4;
  const _Float16* xr = Xh + (size_t)(tile * 16 + n0) * 64;

  v8f acc[4];
  #pragma unroll
  for (int nt = 0; nt < 4; ++nt) acc[nt] = vzero8f();

  #pragma unroll
  for (int ks = 0; ks < 2; ++ks) {
    const int kA = ks * 32 + hi * 8;
    const v16h a = cat16(*(const h8*)(xr + kA), *(const h8*)(xr + kA + 16));
    #pragma unroll
    for (int nt = 0; nt < 4; ++nt) {
      const int col = nt * 16 + n0;
      const _Float16* wp = Ws + col * 64 + ks * 32 + hi * 8;
      const v16h b = cat16(*(const h8*)wp, *(const h8*)(wp + 16));
      acc[nt] = wmma16(a, b, acc[nt]);
    }
  }
  #pragma unroll
  for (int nt = 0; nt < 4; ++nt) {
    const int col = nt * 16 + n0;
    const float bb = bias[col];
    if (MODE == 0) {
      #pragma unroll
      for (int r = 0; r < 8; ++r) {
        const int row = tile * 16 + r + 8 * hi;
        out[(size_t)row * 64 + col] = (_Float16)((acc[nt][r] + bb) * scale);
      }
    } else {
      h8 st;
      #pragma unroll
      for (int r = 0; r < 8; ++r) st[r] = (_Float16)((acc[nt][r] + bb) * scale);
      *(h8*)(out + (size_t)col * N_NUM + tile * 16 + 8 * hi) = st;
    }
  }
}

// ------------------------------------------------------- flash attention
// One wave per (cluster, head, 16-query tile); online softmax over 2048 keys.
__global__ __launch_bounds__(256) void flash_kernel(const _Float16* __restrict__ Qh,
                                                    const _Float16* __restrict__ Kh,
                                                    const _Float16* __restrict__ VTh,
                                                    _Float16* __restrict__ ctxh) {
  __shared__ __align__(16) _Float16 Pt[8][16][16];
  const int w = blockIdx.x * (blockDim.x >> 5) + (threadIdx.x >> 5);
  const int ws = threadIdx.x >> 5;
  const int lane = threadIdx.x & 31;
  const int n0 = lane & 15;
  const int hi = lane >> 4;

  const int c = w >> 9;            // 512 tasks per cluster
  const int rem = w & 511;
  const int h = rem >> 7;          // 4 heads
  const int qt = rem & 127;        // 128 query tiles

  // Q A-fragment (Dh=16 zero-padded to K=32); Q pre-scaled by 1/sqrt(Dh).
  const int qrow = c * G_NUM + qt * 16 + n0;
  const h8 q0 = *(const h8*)(Qh + (size_t)qrow * 64 + h * 16 + 8 * hi);
  const v16h aq = cat16(q0, hzero8());

  v8f o = vzero8f();
  float mrow[8], lrow[8];
  #pragma unroll
  for (int r = 0; r < 8; ++r) { mrow[r] = -1e30f; lrow[r] = 0.f; }

  for (int kt = 0; kt < G_NUM / 16; ++kt) {
    // K B-fragment: lane half hi holds dh 8*hi..8*hi+7 for key n0 in elems 0-7; elems 8-15 pad zero.
    const int key = c * G_NUM + kt * 16 + n0;
    const _Float16* kp = Kh + (size_t)key * 64 + h * 16;
    v16h bk = cat16(*(const h8*)(kp + 8 * hi), hzero8());

    v8f s = wmma16(aq, bk, vzero8f());

    float p[8];
    #pragma unroll
    for (int r = 0; r < 8; ++r) {
      const float v = s[r];
      const float rm = redmax16(v);
      const float nm = fmaxf(mrow[r], rm);
      const float sc = __expf(mrow[r] - nm);
      mrow[r] = nm;
      const float pr = __expf(v - nm);
      lrow[r] = lrow[r] * sc + redsum16(pr);
      o[r] *= sc;
      p[r] = pr;
    }
    // Re-layout P (C/D layout) -> A-fragment layout via per-wave LDS tile.
    #pragma unroll
    for (int r = 0; r < 8; ++r) Pt[ws][r + 8 * hi][n0] = (_Float16)p[r];
    asm volatile("s_wait_dscnt 0" ::: "memory");
    const h8 pl = *(const h8*)&Pt[ws][n0][8 * hi];
    const v16h ap = cat16(pl, hzero8());

    // V B-fragment from transposed V: keys 8*hi..8*hi+7 in elems 0-7, zero pad (dh = n0).
    const _Float16* vp = VTh + (size_t)(h * 16 + n0) * N_NUM + c * G_NUM + kt * 16;
    v16h bv = cat16(*(const h8*)(vp + 8 * hi), hzero8());

    o = wmma16(ap, bv, o);
  }
  #pragma unroll
  for (int r = 0; r < 8; ++r) {
    const int row = c * G_NUM + qt * 16 + r + 8 * hi;
    ctxh[(size_t)row * 64 + h * 16 + n0] = (_Float16)(o[r] / lrow[r]);
  }
}

// ---------------------------------------------- intra-cluster gating scores
__global__ __launch_bounds__(256) void intra_score_kernel(const _Float16* __restrict__ att1,
                                                          const float* __restrict__ Wa1,
                                                          const float* __restrict__ ba1,
                                                          const float* __restrict__ Wa2,
                                                          const float* __restrict__ ba2,
                                                          float* __restrict__ scores) {
  __shared__ float w1s[64 * 32];
  __shared__ float w2s[32];
  const int t = threadIdx.x;
  for (int idx = t; idx < 64 * 32; idx += blockDim.x) w1s[idx] = Wa1[idx];
  if (t < 32) w2s[t] = Wa2[t];
  __syncthreads();

  const int i = blockIdx.x * blockDim.x + t;
  const _Float16* ar = att1 + (size_t)i * 64;
  float hbuf[32];
  #pragma unroll
  for (int j = 0; j < 32; ++j) hbuf[j] = ba1[j];
  for (int d = 0; d < 64; ++d) {
    const float ad = (float)ar[d];
    #pragma unroll
    for (int j = 0; j < 32; ++j) hbuf[j] += ad * w1s[d * 32 + j];
  }
  float s = ba2[0];
  #pragma unroll
  for (int j = 0; j < 32; ++j) s += w2s[j] * tanhf(hbuf[j]);
  scores[i] = s;
}

// ---------------------------------- per-cluster softmax pooling -> feats
__global__ __launch_bounds__(256) void cluster_pool_kernel(const float* __restrict__ scores,
                                                           const _Float16* __restrict__ att1,
                                                           float* __restrict__ wbuf,
                                                           float* __restrict__ feats,
                                                           float* __restrict__ intra_out) {
  __shared__ float red[256];
  __shared__ float smax, ssum;
  const int c = blockIdx.x;
  const int t = threadIdx.x;

  float mx = -1e30f;
  for (int i = t; i < G_NUM; i += 256) mx = fmaxf(mx, scores[c * G_NUM + i]);
  red[t] = mx; __syncthreads();
  for (int s = 128; s > 0; s >>= 1) { if (t < s) red[t] = fmaxf(red[t], red[t + s]); __syncthreads(); }
  if (t == 0) smax = red[0];
  __syncthreads();

  float se = 0.f;
  for (int i = t; i < G_NUM; i += 256) {
    const float e = __expf(scores[c * G_NUM + i] - smax);
    wbuf[c * G_NUM + i] = e;
    se += e;
  }
  red[t] = se; __syncthreads();
  for (int s = 128; s > 0; s >>= 1) { if (t < s) red[t] += red[t + s]; __syncthreads(); }
  if (t == 0) ssum = red[0];
  __syncthreads();

  const float inv = 1.f / ssum;
  for (int i = t; i < G_NUM; i += 256) {
    const float w = wbuf[c * G_NUM + i] * inv;
    wbuf[c * G_NUM + i] = w;
    intra_out[c * G_NUM + i] = w;
  }
  __syncthreads();

  const int d = t & 63, g = t >> 6;
  float acc = 0.f;
  for (int i = g; i < G_NUM; i += 4)
    acc += wbuf[c * G_NUM + i] * (float)att1[(size_t)(c * G_NUM + i) * 64 + d];
  red[t] = acc; __syncthreads();
  if (t < 64) feats[c * 64 + d] = red[t] + red[t + 64] + red[t + 128] + red[t + 192];
}

// ------------------------------------------------- instance classifier
__global__ __launch_bounds__(256) void inst_pred_kernel(const _Float16* __restrict__ att1,
                                                        const float* __restrict__ Wp,
                                                        const float* __restrict__ bp,
                                                        float* __restrict__ out) {
  const int i = blockIdx.x * blockDim.x + threadIdx.x;
  const _Float16* ar = att1 + (size_t)i * 64;
  float z0 = bp[0], z1 = bp[1];
  for (int d = 0; d < 64; ++d) {
    const float a = (float)ar[d];
    z0 += a * Wp[d * 2 + 0];
    z1 += a * Wp[d * 2 + 1];
  }
  const float m = fmaxf(z0, z1);
  const float e0 = __expf(z0 - m), e1 = __expf(z1 - m);
  const float inv = 1.f / (e0 + e1);
  out[i * 2 + 0] = e0 * inv;
  out[i * 2 + 1] = e1 * inv;
}

// ---------------------------------- inter-cluster MHA + pooling + fc6
__global__ __launch_bounds__(64) void final_kernel(const float* __restrict__ feats,
    const float* Wq2, const float* bq2, const float* Wk2, const float* bk2,
    const float* Wv2, const float* bv2, const float* Wo2, const float* bo2,
    const float* Wr1, const float* br1, const float* Wr2, const float* br2,
    const float* Wf1, const float* bf1, const float* Wf2, const float* bf2,
    float* __restrict__ d_out) {
  __shared__ float F[16][64], Qs[16][64], Ks[16][64], Vs[16][64];
  __shared__ float Os[16][64], A2[16][64], RS[16], IW[16], FIN[64];
  const int t = threadIdx.x;
  for (int idx = t; idx < 16 * 64; idx += 64) F[idx >> 6][idx & 63] = feats[idx];
  __syncthreads();

  const int row = t & 15, cg = t >> 4;
  for (int cc = 0; cc < 16; ++cc) {
    const int col = cg * 16 + cc;
    float aq = bq2[col], ak = bk2[col], av = bv2[col];
    for (int d = 0; d < 64; ++d) {
      const float fv = F[row][d];
      aq += fv * Wq2[d * 64 + col];
      ak += fv * Wk2[d * 64 + col];
      av += fv * Wv2[d * 64 + col];
    }
    Qs[row][col] = aq; Ks[row][col] = ak; Vs[row][col] = av;
  }
  __syncthreads();

  const int h = t >> 4;
  float sc[16]; float mx = -1e30f;
  for (int m = 0; m < 16; ++m) {
    float s = 0.f;
    for (int d = 0; d < DH; ++d) s += Qs[row][h * DH + d] * Ks[m][h * DH + d];
    s *= 0.25f;
    sc[m] = s; mx = fmaxf(mx, s);
  }
  float se = 0.f;
  for (int m = 0; m < 16; ++m) { sc[m] = __expf(sc[m] - mx); se += sc[m]; }
  const float inv = 1.f / se;
  for (int d = 0; d < DH; ++d) {
    float ov = 0.f;
    for (int m = 0; m < 16; ++m) ov += sc[m] * Vs[m][h * DH + d];
    Os[row][h * DH + d] = ov * inv;
  }
  __syncthreads();

  for (int cc = 0; cc < 16; ++cc) {
    const int col = cg * 16 + cc;
    float a = bo2[col];
    for (int d = 0; d < 64; ++d) a += Os[row][d] * Wo2[d * 64 + col];
    A2[row][col] = a;
  }
  __syncthreads();

  if (t < 16) {
    float hb[32];
    #pragma unroll
    for (int j = 0; j < 32; ++j) hb[j] = br1[j];
    for (int d = 0; d < 64; ++d) {
      const float ad = A2[t][d];
      #pragma unroll
      for (int j = 0; j < 32; ++j) hb[j] += ad * Wr1[d * 32 + j];
    }
    float s = br2[0];
    #pragma unroll
    for (int j = 0; j < 32; ++j) s += Wr2[j] * tanhf(hb[j]);
    RS[t] = s;
  }
  __syncthreads();

  if (t == 0) {
    float m2 = -1e30f;
    for (int c = 0; c < 16; ++c) m2 = fmaxf(m2, RS[c]);
    float s2 = 0.f;
    for (int c = 0; c < 16; ++c) { IW[c] = __expf(RS[c] - m2); s2 += IW[c]; }
    const float iv = 1.f / s2;
    for (int c = 0; c < 16; ++c) { IW[c] *= iv; d_out[2 + c] = IW[c]; }
  }
  __syncthreads();

  { float s = 0.f;
    for (int c = 0; c < 16; ++c) s += IW[c] * A2[c][t];
    FIN[t] = s; }
  __syncthreads();

  if (t == 0) {
    float hb[32];
    #pragma unroll
    for (int j = 0; j < 32; ++j) {
      float a = bf1[j];
      for (int d = 0; d < 64; ++d) a += FIN[d] * Wf1[d * 32 + j];
      hb[j] = fmaxf(a, 0.f);
    }
    float z0 = bf2[0], z1 = bf2[1];
    #pragma unroll
    for (int j = 0; j < 32; ++j) { z0 += hb[j] * Wf2[j * 2]; z1 += hb[j] * Wf2[j * 2 + 1]; }
    const float m3 = fmaxf(z0, z1);
    const float e0 = __expf(z0 - m3), e1 = __expf(z1 - m3);
    const float iv = 1.f / (e0 + e1);
    d_out[0] = e0 * iv;
    d_out[1] = e1 * iv;
  }
}

// ------------------------------------------------------------- launch
static inline size_t align256(size_t x) { return (x + 255) & ~size_t(255); }

extern "C" void kernel_launch(void* const* d_in, const int* in_sizes, int n_in,
                              void* d_out, int out_size, void* d_ws, size_t ws_size,
                              hipStream_t stream) {
  const float* x       = (const float*)d_in[0];
  const int*   label   = (const int*)d_in[1];
  const int*   barcode = (const int*)d_in[4];
  const float* We = (const float*)d_in[5];  const float* be = (const float*)d_in[6];
  const float* Wq1 = (const float*)d_in[7]; const float* bq1 = (const float*)d_in[8];
  const float* Wk1 = (const float*)d_in[9]; const float* bk1 = (const float*)d_in[10];
  const float* Wv1 = (const float*)d_in[11]; const float* bv1 = (const float*)d_in[12];
  const float* Wo1 = (const float*)d_in[13]; const float* bo1 = (const float*)d_in[14];
  const float* Wa1 = (const float*)d_in[15]; const float* ba1 = (const float*)d_in[16];
  const float* Wa2 = (const float*)d_in[17]; const float* ba2 = (const float*)d_in[18];
  const float* Wq2 = (const float*)d_in[19]; const float* bq2 = (const float*)d_in[20];
  const float* Wk2 = (const float*)d_in[21]; const float* bk2 = (const float*)d_in[22];
  const float* Wv2 = (const float*)d_in[23]; const float* bv2 = (const float*)d_in[24];
  const float* Wo2 = (const float*)d_in[25]; const float* bo2 = (const float*)d_in[26];
  const float* Wr1 = (const float*)d_in[27]; const float* br1 = (const float*)d_in[28];
  const float* Wr2 = (const float*)d_in[29]; const float* br2 = (const float*)d_in[30];
  const float* Wf1 = (const float*)d_in[31]; const float* bf1 = (const float*)d_in[32];
  const float* Wf2 = (const float*)d_in[33]; const float* bf2 = (const float*)d_in[34];
  const float* Wp  = (const float*)d_in[35]; const float* bp  = (const float*)d_in[36];

  float* out = (float*)d_out;
  float* y_out     = out;             // [2]
  float* inter_out = out + 2;         // [16]
  float* intra_out = out + 18;        // [32768]
  float* bcs_out   = out + 18 + N_NUM;        // [32768]
  float* inst_out  = out + 18 + 2 * N_NUM;    // [65536]

  char* ws = (char*)d_ws;
  size_t off = 0;
  auto take = [&](size_t bytes) { void* p = ws + off; off += align256(bytes); return p; };
  int*      ordp   = (int*)take(N_NUM * 4);
  float*    scores = (float*)take(N_NUM * 4);
  float*    wbuf   = (float*)take(N_NUM * 4);
  float*    feats  = (float*)take(C_NUM * 64 * 4);
  _Float16* WeT    = (_Float16*)take((size_t)D_IN * 64 * 2);
  _Float16* Wqt    = (_Float16*)take(4096 * 2);
  _Float16* Wkt    = (_Float16*)take(4096 * 2);
  _Float16* Wvt    = (_Float16*)take(4096 * 2);
  _Float16* Wot    = (_Float16*)take(4096 * 2);
  _Float16* embH   = (_Float16*)take((size_t)N_NUM * 64 * 2);
  _Float16* Qh     = (_Float16*)take((size_t)N_NUM * 64 * 2);
  _Float16* Kh     = (_Float16*)take((size_t)N_NUM * 64 * 2);
  _Float16* VTh    = (_Float16*)take((size_t)N_NUM * 64 * 2);
  _Float16* ctxh   = (_Float16*)take((size_t)N_NUM * 64 * 2);
  _Float16* att1h  = (_Float16*)take((size_t)N_NUM * 64 * 2);
  (void)ws_size; (void)n_in; (void)in_sizes; (void)out_size;

  order_kernel<<<16, 32, 0, stream>>>(label, barcode, ordp, bcs_out);
  prep_kernel<<<256, 256, 0, stream>>>(We, Wq1, Wk1, Wv1, Wo1, WeT, Wqt, Wkt, Wvt, Wot);
  embed_kernel<<<256, 256, 0, stream>>>(x, ordp, be, WeT, embH);
  proj64_kernel<0><<<256, 256, 0, stream>>>(embH, Wqt, bq1, 0.25f, Qh);  // 1/sqrt(Dh) baked in
  proj64_kernel<0><<<256, 256, 0, stream>>>(embH, Wkt, bk1, 1.0f, Kh);
  proj64_kernel<1><<<256, 256, 0, stream>>>(embH, Wvt, bv1, 1.0f, VTh);
  flash_kernel<<<1024, 256, 0, stream>>>(Qh, Kh, VTh, ctxh);
  proj64_kernel<0><<<256, 256, 0, stream>>>(ctxh, Wot, bo1, 1.0f, att1h);
  intra_score_kernel<<<N_NUM / 256, 256, 0, stream>>>(att1h, Wa1, ba1, Wa2, ba2, scores);
  cluster_pool_kernel<<<16, 256, 0, stream>>>(scores, att1h, wbuf, feats, intra_out);
  inst_pred_kernel<<<N_NUM / 256, 256, 0, stream>>>(att1h, Wp, bp, inst_out);
  final_kernel<<<1, 64, 0, stream>>>(feats, Wq2, bq2, Wk2, bk2, Wv2, bv2, Wo2, bo2,
                                     Wr1, br1, Wr2, br2, Wf1, bf1, Wf2, bf2, y_out);
}
